// FCSwitchedVAE_44985487458670
// MI455X (gfx1250) — compile-verified, matrix operand order checked by reading, not run on hardware
//
#include <hip/hip_runtime.h>
#include <hip/hip_bf16.h>

// Problem constants (from reference): B,C,D,SM,NB,NS = 512,1,1024,256,8,4
#define BATCH 512
#define DIMD  1024
#define SMDIM 256
#define NBLK  8
#define NSTG  4
#define LDW   20   // LDS row pitch (floats) for staged weight tiles: conflict-free

typedef __attribute__((ext_vector_type(16))) __bf16 v16bf;
typedef __attribute__((ext_vector_type(8)))  float  v8f;

// ---------------------------------------------------------------------------
// WMMA fragment helpers (layouts per CDNA5 ISA):
// A 16x32 bf16: lane l -> m=l&15, half=l>>4;
//   elems 0..7 = A[m][k0+half*8+e], elems 8..15 = A[m][k0+16+half*8+e]
// B mirrored over N-lanes.
// ---------------------------------------------------------------------------
__device__ __forceinline__ v16bf load_bfrag_g(const float* __restrict__ wp, int ld,
                                              int k0, int half, int lm) {
  v16bf b;
#pragma unroll
  for (int e = 0; e < 8; ++e) {
    b[e]     = (__bf16)wp[(size_t)(k0 + half * 8 + e) * ld + lm];
    b[e + 8] = (__bf16)wp[(size_t)(k0 + 16 + half * 8 + e) * ld + lm];
  }
  return b;
}

__device__ __forceinline__ v16bf load_bfrag_lds(const float* __restrict__ lw,
                                                int half, int lm) {
  v16bf b;
#pragma unroll
  for (int e = 0; e < 8; ++e) {
    b[e]     = (__bf16)lw[(half * 8 + e) * LDW + lm];
    b[e + 8] = (__bf16)lw[(16 + half * 8 + e) * LDW + lm];
  }
  return b;
}

__device__ __forceinline__ v16bf load_afrag_row(const float* __restrict__ arow,
                                                int k0, int half, bool relu) {
  float av[16];
  *(float4*)&av[0]  = *(const float4*)(arow + k0 + half * 8);
  *(float4*)&av[4]  = *(const float4*)(arow + k0 + half * 8 + 4);
  *(float4*)&av[8]  = *(const float4*)(arow + k0 + 16 + half * 8);
  *(float4*)&av[12] = *(const float4*)(arow + k0 + 16 + half * 8 + 4);
  v16bf a;
#pragma unroll
  for (int e = 0; e < 16; ++e) a[e] = (__bf16)(relu ? fmaxf(av[e], 0.0f) : av[e]);
  return a;
}

// Async-stage one 32(K) x 16(N) f32 weight tile into LDS (ASYNCcnt-tracked).
// 32 lanes x 16B per issue = 512B; 4 issues cover the 2KB tile.
// lane -> (row = lane>>2 (+8i), chunk = lane&3); rows are 64B, 16B-aligned.
__device__ __forceinline__ void async_load_wtile(const float* __restrict__ wsrc, int ld,
                                                 unsigned ldsbase, int lane) {
  int r = lane >> 2, c = lane & 3;
#pragma unroll
  for (int i = 0; i < 4; ++i) {
    const float* g = wsrc + (size_t)(r + 8 * i) * ld + c * 4;
    unsigned l = ldsbase + (unsigned)(((r + 8 * i) * LDW + c * 4) * 4);
    asm volatile("global_load_async_to_lds_b128 %0, %1, off" :: "v"(l), "v"(g) : "memory");
  }
}

// ---------------------------------------------------------------------------
// small utility kernels
// ---------------------------------------------------------------------------
__global__ void zero_counts_kernel(int* __restrict__ counts, int n) {
  int i = blockIdx.x * blockDim.x + threadIdx.x;
  if (i < n) counts[i] = 0;
}

__global__ void relu_inplace_kernel(float* __restrict__ x, int n) {
  int i = blockIdx.x * blockDim.x + threadIdx.x;
  if (i < n) x[i] = fmaxf(x[i], 0.0f);
}

// pack conv weight (OC,IC,4,4) -> wp[k][oc], k = ic*16+ky*4+kx (== w inner flat)
__global__ void pack_conv_w_kernel(const float* __restrict__ w, float* __restrict__ wp,
                                   int IC, int OC) {
  int idx = blockIdx.x * blockDim.x + threadIdx.x;
  int total = IC * 16 * OC;
  if (idx >= total) return;
  int oc = idx % OC, k = idx / OC;
  wp[idx] = w[(size_t)oc * IC * 16 + k];
}

// pack deconv weight (IC,OC,4,4) -> wp[par][k][oc], par=(py,px), k=ic*4+jy*2+jx,
// ky = 1-py+2*jy, kx = 1-px+2*jx  (fixed 2x2 valid-tap set per output parity)
__global__ void pack_deconv_w_kernel(const float* __restrict__ w, float* __restrict__ wp,
                                     int IC, int OC) {
  int idx = blockIdx.x * blockDim.x + threadIdx.x;
  int K = IC * 4;
  int total = 4 * K * OC;
  if (idx >= total) return;
  int oc = idx % OC; int r = idx / OC;
  int k = r % K;     int par = r / K;
  int ic = k >> 2, jy = (k >> 1) & 1, jx = k & 1;
  int py = par >> 1, px = par & 1;
  int ky = 1 - py + 2 * jy, kx = 1 - px + 2 * jx;
  wp[idx] = w[((size_t)(ic * OC + oc) * 4 + ky) * 4 + kx];
}

// ---------------------------------------------------------------------------
// Direct conv, stride 2, pad 1, k=4 (conv1 only: C=1 -> K=16, no WMMA tile)
// ---------------------------------------------------------------------------
template <int IC, int OC, int IH, int IW, bool RELU>
__global__ void conv_s2_kernel(const float* __restrict__ x, const float* __restrict__ w,
                               const float* __restrict__ bias, float* __restrict__ y) {
  constexpr int OH = IH / 2, OW = IW / 2;
  constexpr int total = BATCH * OC * OH * OW;
  int idx = blockIdx.x * blockDim.x + threadIdx.x;
  if (idx >= total) return;
  int ox = idx % OW; int t = idx / OW;
  int oy = t % OH;   t /= OH;
  int oc = t % OC;   int b = t / OC;
  float acc = bias[oc];
  for (int ic = 0; ic < IC; ++ic) {
    const float* xp = x + ((size_t)(b * IC + ic) * IH) * IW;
    const float* wpt = w + (size_t)(oc * IC + ic) * 16;
#pragma unroll
    for (int ky = 0; ky < 4; ++ky) {
      int iy = oy * 2 - 1 + ky;
      if ((unsigned)iy >= (unsigned)IH) continue;
#pragma unroll
      for (int kx = 0; kx < 4; ++kx) {
        int ix = ox * 2 - 1 + kx;
        if ((unsigned)ix >= (unsigned)IW) continue;
        acc += xp[iy * IW + ix] * wpt[ky * 4 + kx];
      }
    }
  }
  if (RELU) acc = fmaxf(acc, 0.0f);
  y[idx] = acc;
}

// Direct transposed conv (dec4 only: OC=1, no WMMA N-tile)
template <int IC, int OC, int IH, int IW, bool RELU>
__global__ void deconv_direct_kernel(const float* __restrict__ x, const float* __restrict__ w,
                                     const float* __restrict__ bias, float* __restrict__ y) {
  constexpr int OH = IH * 2, OW = IW * 2;
  constexpr int total = BATCH * OC * OH * OW;
  int idx = blockIdx.x * blockDim.x + threadIdx.x;
  if (idx >= total) return;
  int ox = idx % OW; int t = idx / OW;
  int oy = t % OH;   t /= OH;
  int oc = t % OC;   int b = t / OC;
  float acc = bias[oc];
#pragma unroll
  for (int ky = 0; ky < 4; ++ky) {
    int ty = oy + 1 - ky;
    if (ty < 0 || (ty & 1)) continue;
    int iy = ty >> 1;
    if (iy >= IH) continue;
#pragma unroll
    for (int kx = 0; kx < 4; ++kx) {
      int tx = ox + 1 - kx;
      if (tx < 0 || (tx & 1)) continue;
      int ix = tx >> 1;
      if (ix >= IW) continue;
      for (int ic = 0; ic < IC; ++ic) {
        acc += x[((size_t)(b * IC + ic) * IH + iy) * IW + ix] *
               w[((size_t)(ic * OC + oc) * 4 + ky) * 4 + kx];
      }
    }
  }
  if (RELU) acc = fmaxf(acc, 0.0f);
  y[idx] = acc;
}

// ---------------------------------------------------------------------------
// Implicit-GEMM WMMA conv (stride 2, pad 1, k=4): M=B*OH*OW, N=OC, K=IC*16.
// One wave = 64(M) x 16(N); B fragment reused by 4 WMMAs. Compile-time dims.
// ---------------------------------------------------------------------------
template <int IC, int OC, int IH, int IW, bool RELU>
__global__ __launch_bounds__(32)
void conv_wmma_kernel(const float* __restrict__ x, const float* __restrict__ wp,
                      const float* __restrict__ bias, float* __restrict__ y) {
  constexpr int OH = IH / 2, OW = IW / 2, K = IC * 16;
  const int m0 = blockIdx.y * 64, col0 = blockIdx.x * 16;
  const int lane = threadIdx.x, half = lane >> 4, lm = lane & 15;
  const float* __restrict__ wpc = wp + col0;
  int bv[4], oyv[4], oxv[4];
#pragma unroll
  for (int t = 0; t < 4; ++t) {
    int m = m0 + t * 16 + lm;
    oxv[t] = m % OW; int r = m / OW;
    oyv[t] = r % OH; bv[t] = r / OH;
  }
  v8f acc[4] = {};
  for (int k0 = 0; k0 < K; k0 += 32) {
    v16bf bf = load_bfrag_g(wpc, OC, k0, half, lm);
#pragma unroll
    for (int t = 0; t < 4; ++t) {
      v16bf a;
#pragma unroll
      for (int e = 0; e < 16; ++e) {
        int k = k0 + (e < 8 ? half * 8 + e : 16 + half * 8 + (e - 8));
        int ic = k >> 4, ky = (k >> 2) & 3, kx = k & 3;
        int iy = oyv[t] * 2 - 1 + ky, ix = oxv[t] * 2 - 1 + kx;
        float v = 0.0f;
        if ((unsigned)iy < (unsigned)IH && (unsigned)ix < (unsigned)IW)
          v = x[((size_t)(bv[t] * IC + ic) * IH + iy) * IW + ix];
        a[e] = (__bf16)v;
      }
      acc[t] = __builtin_amdgcn_wmma_f32_16x16x32_bf16(false, a, false, bf,
                                                       (short)0, acc[t], false, false);
    }
  }
  const float bs = bias[col0 + lm];
#pragma unroll
  for (int t = 0; t < 4; ++t) {
#pragma unroll
    for (int v = 0; v < 8; ++v) {
      int m = m0 + t * 16 + v + 8 * half;      // C/D layout: m = v + 8*half
      int ox = m % OW; int r = m / OW;
      int oy = r % OH; int b = r / OH;
      float o = acc[t][v] + bs;
      if (RELU) o = fmaxf(o, 0.0f);
      y[((size_t)(b * OC + col0 + lm) * OH + oy) * OW + ox] = o;
    }
  }
}

// ---------------------------------------------------------------------------
// Implicit-GEMM WMMA transposed conv, parity-decomposed (blockIdx.z=(py,px)):
// iy = oy2 + py - jy, ix = ox2 + px - jx. M=B*IH*IW, N=OC, K=IC*4.
// ---------------------------------------------------------------------------
template <int IC, int OC, int IH, int IW, bool RELU>
__global__ __launch_bounds__(32)
void deconv_wmma_kernel(const float* __restrict__ x, const float* __restrict__ wp,
                        const float* __restrict__ bias, float* __restrict__ y) {
  constexpr int K = IC * 4, OH = IH * 2, OW = IW * 2;
  const int par = blockIdx.z, py = par >> 1, px = par & 1;
  const int m0 = blockIdx.y * 64, col0 = blockIdx.x * 16;
  const int lane = threadIdx.x, half = lane >> 4, lm = lane & 15;
  const float* __restrict__ wpc = wp + (size_t)par * K * OC + col0;
  int bv[4], oyv[4], oxv[4];
#pragma unroll
  for (int t = 0; t < 4; ++t) {
    int m = m0 + t * 16 + lm;
    oxv[t] = m % IW; int r = m / IW;
    oyv[t] = r % IH; bv[t] = r / IH;
  }
  v8f acc[4] = {};
  for (int k0 = 0; k0 < K; k0 += 32) {
    v16bf bf = load_bfrag_g(wpc, OC, k0, half, lm);
#pragma unroll
    for (int t = 0; t < 4; ++t) {
      v16bf a;
#pragma unroll
      for (int e = 0; e < 16; ++e) {
        int k = k0 + (e < 8 ? half * 8 + e : 16 + half * 8 + (e - 8));
        int ic = k >> 2, jy = (k >> 1) & 1, jx = k & 1;
        int iy = oyv[t] + py - jy, ix = oxv[t] + px - jx;
        float v = 0.0f;
        if ((unsigned)iy < (unsigned)IH && (unsigned)ix < (unsigned)IW)
          v = x[((size_t)(bv[t] * IC + ic) * IH + iy) * IW + ix];
        a[e] = (__bf16)v;
      }
      acc[t] = __builtin_amdgcn_wmma_f32_16x16x32_bf16(false, a, false, bf,
                                                       (short)0, acc[t], false, false);
    }
  }
  const float bs = bias[col0 + lm];
#pragma unroll
  for (int t = 0; t < 4; ++t) {
#pragma unroll
    for (int v = 0; v < 8; ++v) {
      int m = m0 + t * 16 + v + 8 * half;
      int ox2 = m % IW; int r = m / IW;
      int oy2 = r % IH; int b = r / IH;
      float o = acc[t][v] + bs;
      if (RELU) o = fmaxf(o, 0.0f);
      y[((size_t)(b * OC + col0 + lm) * OH + oy2 * 2 + py) * OW + ox2 * 2 + px] = o;
    }
  }
}

// ---------------------------------------------------------------------------
// Router: ctrl = relu(h) @ sw_w + sw_b (24 outs), Gumbel argmax -> expert id,
// z_sel; atomic-append row into per-expert list (order-invariant result).
// ---------------------------------------------------------------------------
__global__ __launch_bounds__(256)
void router_kernel(const float* __restrict__ H, const float* __restrict__ SWW,
                   const float* __restrict__ SWB, const float* __restrict__ GU,
                   const float* __restrict__ ZN, int* __restrict__ rows,
                   int* __restrict__ counts, float* __restrict__ zsel) {
  __shared__ float sred[256];
  __shared__ float sctrl[24];
  const int b = blockIdx.x, t = threadIdx.x;
  float4 hv = ((const float4*)(H + (size_t)b * DIMD))[t];
  hv.x = fmaxf(hv.x, 0.f); hv.y = fmaxf(hv.y, 0.f);
  hv.z = fmaxf(hv.z, 0.f); hv.w = fmaxf(hv.w, 0.f);
  const int d0 = t * 4;
  for (int j = 0; j < 24; ++j) {
    float p = hv.x * SWW[(d0 + 0) * 24 + j] + hv.y * SWW[(d0 + 1) * 24 + j] +
              hv.z * SWW[(d0 + 2) * 24 + j] + hv.w * SWW[(d0 + 3) * 24 + j];
    sred[t] = p; __syncthreads();
    for (int s = 128; s > 0; s >>= 1) {
      if (t < s) sred[t] += sred[t + s];
      __syncthreads();
    }
    if (t == 0) sctrl[j] = sred[0] + SWB[j];
    __syncthreads();
  }
  if (t == 0) {
    int best = 0; float bvv = -3.4e38f;
    for (int j = 0; j < NBLK; ++j) {
      float u = GU[b * NBLK + j];
      float g = -logf(-logf(u + 1e-12f) + 1e-20f);
      float v = sctrl[j] + g;
      if (v > bvv) { bvv = v; best = j; }  // strict > : first max wins (jnp.argmax)
    }
    float zm = sctrl[8 + best], zl = sctrl[16 + best];
    zsel[b] = ZN[b * NBLK + best] * expf(zl * 0.5f) + zm;
    int pos = atomicAdd(&counts[best], 1);
    rows[best * BATCH + pos] = b;
  }
}

// ---------------------------------------------------------------------------
// Mid block: z2_mean/logvar (D->10), reparam, d = z2 @ fc_latent (10->D)
// ---------------------------------------------------------------------------
__global__ __launch_bounds__(256)
void mid_kernel(const float* __restrict__ H,
                const float* __restrict__ MW, const float* __restrict__ MB,
                const float* __restrict__ LW, const float* __restrict__ LB,
                const float* __restrict__ FW, const float* __restrict__ FB,
                const float* __restrict__ Z2N, float* __restrict__ DOUT) {
  __shared__ float sred[256];
  __shared__ float sm[10], sl[10], sz2[10];
  const int b = blockIdx.x, t = threadIdx.x;
  float4 hv = ((const float4*)(H + (size_t)b * DIMD))[t];
  hv.x = fmaxf(hv.x, 0.f); hv.y = fmaxf(hv.y, 0.f);
  hv.z = fmaxf(hv.z, 0.f); hv.w = fmaxf(hv.w, 0.f);
  const int d0 = t * 4;
  for (int j = 0; j < 10; ++j) {
    float p = hv.x * MW[(d0 + 0) * 10 + j] + hv.y * MW[(d0 + 1) * 10 + j] +
              hv.z * MW[(d0 + 2) * 10 + j] + hv.w * MW[(d0 + 3) * 10 + j];
    sred[t] = p; __syncthreads();
    for (int s = 128; s > 0; s >>= 1) { if (t < s) sred[t] += sred[t + s]; __syncthreads(); }
    if (t == 0) sm[j] = sred[0] + MB[j];
    __syncthreads();
    p = hv.x * LW[(d0 + 0) * 10 + j] + hv.y * LW[(d0 + 1) * 10 + j] +
        hv.z * LW[(d0 + 2) * 10 + j] + hv.w * LW[(d0 + 3) * 10 + j];
    sred[t] = p; __syncthreads();
    for (int s = 128; s > 0; s >>= 1) { if (t < s) sred[t] += sred[t + s]; __syncthreads(); }
    if (t == 0) sl[j] = sred[0] + LB[j];
    __syncthreads();
  }
  if (t < 10) sz2[t] = Z2N[b * 10 + t] * expf(sl[t] * 0.5f) + sm[t];
  __syncthreads();
#pragma unroll
  for (int q = 0; q < 4; ++q) {
    int col = d0 + q;
    float acc = FB[col];
#pragma unroll
    for (int p = 0; p < 10; ++p) acc += sz2[p] * FW[p * DIMD + col];
    DOUT[(size_t)b * DIMD + col] = acc;
  }
}

// ---------------------------------------------------------------------------
// MoE expert GEMM 1: eh = relu( gather(relu(X)) @ W1[n] + b1[n] )
// 64(M) x 16(N) per wave; weight tile double-buffered in LDS via
// global_load_async_to_lds_b128 (+ s_wait_asynccnt) to hide fetch latency.
// ---------------------------------------------------------------------------
__global__ __launch_bounds__(32)
void moe_gemm1(const float* __restrict__ X,     // BATCH x DIMD (relu on load)
               const float* __restrict__ W,     // NBLK x DIMD x SMDIM
               const float* __restrict__ bias,  // NBLK x SMDIM
               const int* __restrict__ rows,    // NBLK x BATCH
               const int* __restrict__ counts,  // NBLK
               float* __restrict__ Y)           // NBLK x BATCH x SMDIM
{
  __shared__ __align__(16) float ldsw[2][32 * LDW];
  const int n = blockIdx.z;
  const int cnt = counts[n];
  const int m0 = blockIdx.y * 64;
  if (m0 >= cnt) return;                 // uniform: EXEC all-ones at WMMA
  const int col0 = blockIdx.x * 16;
  const int lane = threadIdx.x, half = lane >> 4, lm = lane & 15;
  const float* __restrict__ arow[4];
#pragma unroll
  for (int t = 0; t < 4; ++t) {
    int ri = m0 + t * 16 + lm;
    if (ri >= cnt) ri = cnt - 1;         // clamped rows: results discarded
    arow[t] = X + (size_t)rows[n * BATCH + ri] * DIMD;
  }
  const float* __restrict__ wmat = W + (size_t)n * DIMD * SMDIM + col0;
  const unsigned lb0 = (unsigned)(uintptr_t)&ldsw[0][0];
  const unsigned lb1 = (unsigned)(uintptr_t)&ldsw[1][0];
  async_load_wtile(wmat, SMDIM, lb0, lane);
  v8f acc[4] = {};
  int buf = 0;
  for (int k0 = 0; k0 < DIMD; k0 += 32, buf ^= 1) {
    if (k0 + 32 < DIMD) {
      async_load_wtile(wmat + (size_t)(k0 + 32) * SMDIM, SMDIM, buf ? lb0 : lb1, lane);
      asm volatile("s_wait_asynccnt 0x4" ::: "memory");
    } else {
      asm volatile("s_wait_asynccnt 0x0" ::: "memory");
    }
    const float* lw = buf ? &ldsw[1][0] : &ldsw[0][0];
    v16bf bf = load_bfrag_lds(lw, half, lm);
#pragma unroll
    for (int t = 0; t < 4; ++t) {
      v16bf a = load_afrag_row(arow[t], k0, half, true);
      acc[t] = __builtin_amdgcn_wmma_f32_16x16x32_bf16(false, a, false, bf,
                                                       (short)0, acc[t], false, false);
    }
  }
  const float bs = bias[n * SMDIM + col0 + lm];
#pragma unroll
  for (int t = 0; t < 4; ++t)
#pragma unroll
    for (int v = 0; v < 8; ++v) {
      int i = m0 + t * 16 + v + 8 * half;
      Y[((size_t)n * BATCH + i) * SMDIM + col0 + lm] = fmaxf(acc[t][v] + bs, 0.0f);
    }
}

// ---------------------------------------------------------------------------
// MoE expert GEMM 2 with fused scatter epilogue:
//   H[rows[n][i], col] += (eh[i] @ W2[n] + b2[n])[col] * zsel[rows[n][i]]
// ---------------------------------------------------------------------------
__global__ __launch_bounds__(32)
void moe_gemm2(const float* __restrict__ EH,    // NBLK x BATCH x SMDIM (relu'd)
               const float* __restrict__ W,     // NBLK x SMDIM x DIMD
               const float* __restrict__ bias,  // NBLK x DIMD
               const int* __restrict__ rows,
               const int* __restrict__ counts,
               const float* __restrict__ zsel,  // BATCH
               float* __restrict__ H)           // BATCH x DIMD (accumulated)
{
  __shared__ __align__(16) float ldsw[2][32 * LDW];
  const int n = blockIdx.z;
  const int cnt = counts[n];
  const int m0 = blockIdx.y * 64;
  if (m0 >= cnt) return;
  const int col0 = blockIdx.x * 16;
  const int lane = threadIdx.x, half = lane >> 4, lm = lane & 15;
  const float* __restrict__ abase = EH + ((size_t)n * BATCH + m0 + lm) * SMDIM;
  const float* __restrict__ wmat = W + (size_t)n * SMDIM * DIMD + col0;
  const unsigned lb0 = (unsigned)(uintptr_t)&ldsw[0][0];
  const unsigned lb1 = (unsigned)(uintptr_t)&ldsw[1][0];
  async_load_wtile(wmat, DIMD, lb0, lane);
  v8f acc[4] = {};
  int buf = 0;
  for (int k0 = 0; k0 < SMDIM; k0 += 32, buf ^= 1) {
    if (k0 + 32 < SMDIM) {
      async_load_wtile(wmat + (size_t)(k0 + 32) * DIMD, DIMD, buf ? lb0 : lb1, lane);
      asm volatile("s_wait_asynccnt 0x4" ::: "memory");
    } else {
      asm volatile("s_wait_asynccnt 0x0" ::: "memory");
    }
    const float* lw = buf ? &ldsw[1][0] : &ldsw[0][0];
    v16bf bf = load_bfrag_lds(lw, half, lm);
#pragma unroll
    for (int t = 0; t < 4; ++t) {
      v16bf a = load_afrag_row(abase + (size_t)t * 16 * SMDIM, k0, half, false);
      acc[t] = __builtin_amdgcn_wmma_f32_16x16x32_bf16(false, a, false, bf,
                                                       (short)0, acc[t], false, false);
    }
  }
  const float bs = bias[(size_t)n * DIMD + col0 + lm];
#pragma unroll
  for (int t = 0; t < 4; ++t)
#pragma unroll
    for (int v = 0; v < 8; ++v) {
      int i = m0 + t * 16 + v + 8 * half;
      if (i < cnt) {
        int b = rows[n * BATCH + i];
        H[(size_t)b * DIMD + col0 + lm] += (acc[t][v] + bs) * zsel[b];
      }
    }
}

// ---------------------------------------------------------------------------
// host-side launch
// ---------------------------------------------------------------------------
static inline int cdiv(int a, int b) { return (a + b - 1) / b; }

extern "C" void kernel_launch(void* const* d_in, const int* in_sizes, int n_in,
                              void* d_out, int out_size, void* d_ws, size_t ws_size,
                              hipStream_t stream) {
  (void)in_sizes; (void)n_in; (void)out_size; (void)ws_size;
  const float* x         = (const float*)d_in[0];
  const float* conv1_w   = (const float*)d_in[1];
  const float* conv1_b   = (const float*)d_in[2];
  const float* conv2_w   = (const float*)d_in[3];
  const float* conv2_b   = (const float*)d_in[4];
  const float* conv3_w   = (const float*)d_in[5];
  const float* conv3_b   = (const float*)d_in[6];
  const float* conv4_w   = (const float*)d_in[7];
  const float* conv4_b   = (const float*)d_in[8];
  const float* enc_sw_w  = (const float*)d_in[9];
  const float* enc_sw_b  = (const float*)d_in[10];
  const float* enc_w1    = (const float*)d_in[11];
  const float* enc_b1    = (const float*)d_in[12];
  const float* enc_w2    = (const float*)d_in[13];
  const float* enc_b2    = (const float*)d_in[14];
  const float* fc_mean_w = (const float*)d_in[15];
  const float* fc_mean_b = (const float*)d_in[16];
  const float* fc_lv_w   = (const float*)d_in[17];
  const float* fc_lv_b   = (const float*)d_in[18];
  const float* fc_lat_w  = (const float*)d_in[19];
  const float* fc_lat_b  = (const float*)d_in[20];
  const float* dec_w1    = (const float*)d_in[21];
  const float* dec_b1    = (const float*)d_in[22];
  const float* dec_w2    = (const float*)d_in[23];
  const float* dec_b2    = (const float*)d_in[24];
  const float* dec1_w    = (const float*)d_in[25];
  const float* dec1_b    = (const float*)d_in[26];
  const float* dec2_w    = (const float*)d_in[27];
  const float* dec2_b    = (const float*)d_in[28];
  const float* dec3_w    = (const float*)d_in[29];
  const float* dec3_b    = (const float*)d_in[30];
  const float* dec4_w    = (const float*)d_in[31];
  const float* dec4_b    = (const float*)d_in[32];
  const float* gumbel_u  = (const float*)d_in[33];
  const float* z_noise   = (const float*)d_in[34];
  const float* z2_noise  = (const float*)d_in[35];
  float* out = (float*)d_out;

  // workspace layout (256B aligned); big buffers reused enc<->dec
  char* ws = (char*)d_ws;
  size_t off = 0;
  auto take = [&](size_t bytes) { size_t o = off; off += (bytes + 255) & ~(size_t)255; return o; };
  float* h      = (float*)(ws + take((size_t)BATCH * DIMD * 4));          // 2 MB
  float* dbuf   = (float*)(ws + take((size_t)BATCH * DIMD * 4));          // 2 MB
  float* eh     = (float*)(ws + take((size_t)NBLK * BATCH * SMDIM * 4));  // 4 MB
  int*   rows   = (int*)  (ws + take((size_t)NSTG * NBLK * BATCH * 4));
  int*   counts = (int*)  (ws + take((size_t)NSTG * NBLK * 4));
  float* zsel   = (float*)(ws + take((size_t)NSTG * BATCH * 4));
  float* wbuf   = (float*)(ws + take((size_t)512 * 1024));                // packed conv weights
  float* c1     = (float*)(ws + take((size_t)BATCH * 32 * 32 * 32 * 4));  // 64 MB (also dec3 out)
  float* c2     = (float*)(ws + take((size_t)BATCH * 32 * 16 * 16 * 4));  // 16 MB (also dec2 out)
  float* c3     = (float*)(ws + take((size_t)BATCH * 64 * 8 * 8 * 4));    //  8 MB (also dec1 out)

  zero_counts_kernel<<<1, 64, 0, stream>>>(counts, NSTG * NBLK);

  // --- conv encoder: conv1 direct (K=16), conv2..4 implicit-GEMM WMMA ---
  conv_s2_kernel<1, 32, 64, 64, true><<<cdiv(BATCH * 32 * 32 * 32, 256), 256, 0, stream>>>(x, conv1_w, conv1_b, c1);

  pack_conv_w_kernel<<<cdiv(32 * 16 * 32, 256), 256, 0, stream>>>(conv2_w, wbuf, 32, 32);
  conv_wmma_kernel<32, 32, 32, 32, true><<<dim3(2, 2048), 32, 0, stream>>>(c1, wbuf, conv2_b, c2);
  pack_conv_w_kernel<<<cdiv(32 * 16 * 64, 256), 256, 0, stream>>>(conv3_w, wbuf, 32, 64);
  conv_wmma_kernel<32, 64, 16, 16, true><<<dim3(4, 512), 32, 0, stream>>>(c2, wbuf, conv3_b, c3);
  pack_conv_w_kernel<<<cdiv(64 * 16 * 64, 256), 256, 0, stream>>>(conv4_w, wbuf, 64, 64);
  conv_wmma_kernel<64, 64, 8, 8, false><<<dim3(4, 128), 32, 0, stream>>>(c3, wbuf, conv4_b, h);

  // --- encoder MoE stages (route once, compute only the selected expert) ---
  dim3 g1(SMDIM / 16, BATCH / 64, NBLK);
  dim3 g2(DIMD / 16, BATCH / 64, NBLK);
  for (int s = 0; s < NSTG; ++s) {
    int*   rows_s = rows + s * NBLK * BATCH;
    int*   cnt_s  = counts + s * NBLK;
    float* zs_s   = zsel + s * BATCH;
    router_kernel<<<BATCH, 256, 0, stream>>>(
        h, enc_sw_w + (size_t)s * DIMD * 24, enc_sw_b + s * 24,
        gumbel_u + (size_t)s * BATCH * NBLK, z_noise + (size_t)s * BATCH * NBLK,
        rows_s, cnt_s, zs_s);
    moe_gemm1<<<g1, 32, 0, stream>>>(
        h, enc_w1 + (size_t)s * NBLK * DIMD * SMDIM, enc_b1 + (size_t)s * NBLK * SMDIM,
        rows_s, cnt_s, eh);
    moe_gemm2<<<g2, 32, 0, stream>>>(
        eh, enc_w2 + (size_t)s * NBLK * SMDIM * DIMD, enc_b2 + (size_t)s * NBLK * DIMD,
        rows_s, cnt_s, zs_s, h);
  }

  // --- latent bottleneck ---
  mid_kernel<<<BATCH, 256, 0, stream>>>(h, fc_mean_w, fc_mean_b, fc_lv_w, fc_lv_b,
                                        fc_lat_w, fc_lat_b, z2_noise, dbuf);

  // --- decoder MoE stages (reuse encoder routing + z_sel) ---
  for (int s = 0; s < NSTG; ++s) {
    int*   rows_s = rows + s * NBLK * BATCH;
    int*   cnt_s  = counts + s * NBLK;
    float* zs_s   = zsel + s * BATCH;
    moe_gemm1<<<g1, 32, 0, stream>>>(
        dbuf, dec_w1 + (size_t)s * NBLK * DIMD * SMDIM, dec_b1 + (size_t)s * NBLK * SMDIM,
        rows_s, cnt_s, eh);
    moe_gemm2<<<g2, 32, 0, stream>>>(
        eh, dec_w2 + (size_t)s * NBLK * SMDIM * DIMD, dec_b2 + (size_t)s * NBLK * DIMD,
        rows_s, cnt_s, zs_s, dbuf);
  }

  // --- deconv decoder: dec1..3 parity-decomposed WMMA, dec4 direct (OC=1) ---
  relu_inplace_kernel<<<cdiv(BATCH * DIMD, 256), 256, 0, stream>>>(dbuf, BATCH * DIMD);

  pack_deconv_w_kernel<<<cdiv(4 * 64 * 4 * 64, 256), 256, 0, stream>>>(dec1_w, wbuf, 64, 64);
  deconv_wmma_kernel<64, 64, 4, 4, true><<<dim3(4, 128, 4), 32, 0, stream>>>(dbuf, wbuf, dec1_b, c3);
  pack_deconv_w_kernel<<<cdiv(4 * 64 * 4 * 32, 256), 256, 0, stream>>>(dec2_w, wbuf, 64, 32);
  deconv_wmma_kernel<64, 32, 8, 8, true><<<dim3(2, 512, 4), 32, 0, stream>>>(c3, wbuf, dec2_b, c2);
  pack_deconv_w_kernel<<<cdiv(4 * 32 * 4 * 32, 256), 256, 0, stream>>>(dec3_w, wbuf, 32, 32);
  deconv_wmma_kernel<32, 32, 16, 16, true><<<dim3(2, 2048, 4), 32, 0, stream>>>(c2, wbuf, dec3_b, c1);

  deconv_direct_kernel<32, 1, 32, 32, false><<<cdiv(BATCH * 1 * 64 * 64, 256), 256, 0, stream>>>(c1, dec4_w, dec4_b, out);
}
